// DGCNN_patch_semseg_48670569398534
// MI455X (gfx1250) — matrix operand order repaired and checked by reading, not compiled, not run on hardware
//
#include <hip/hip_runtime.h>
#include <hip/hip_bf16.h>
#include <math.h>

// ---------------------------------------------------------------------------
// MI455X (gfx1250) implementation notes:
//  * All matmuls (conv1d/conv2d-1x1/linear/attention) go through one bf16 WMMA
//    GEMM kernel: f32 global -> bf16 LDS tiles -> v_wmma_f32_16x16x32_bf16,
//    f32 accumulate.  Compute-bound workload (~0.4 TFLOP vs ~17us of HBM time
//    at 23.3 TB/s), so the 16-bit matrix pipe is the right ceiling.
//  * wave32: 256-thread block = 8 waves arranged 2x4 over a 128x128 tile;
//    each wave owns 4x2 16x16 accumulators (64 VGPRs of f32 accum).
//  * Fast-path staging: unguarded global_load_b128 + packed bf16 ds_store_b64
//    when the tile is fully in range (covers all large-K GEMMs); guarded
//    scalar path only for edge tiles / tiny-K (K=3,6,16) GEMMs.
//  * __builtin_prefetch on the next K tile emits global_prefetch_b8.
// ---------------------------------------------------------------------------

typedef __bf16 bf16_t;
typedef __bf16 v8bf  __attribute__((ext_vector_type(8)));
typedef __bf16 v16bf __attribute__((ext_vector_type(16)));
typedef float  v8f   __attribute__((ext_vector_type(8)));

static constexpr int B_  = 4;
static constexpr int N_  = 2048;
static constexpr int BN_ = B_ * N_;
static constexpr int KNN = 20;

#define EPI_NONE       0
#define EPI_BIAS       1
#define EPI_BIAS_RELU  2
#define EPI_BIAS_LRELU 3
#define EPI_SS         4

__device__ __forceinline__ unsigned bf1(float f) {
  unsigned u = __float_as_uint(f);
  return (u + 0x7FFFu + ((u >> 16) & 1u)) >> 16;   // RNE f32 -> bf16 bits
}
__device__ __forceinline__ unsigned pack2bf(float a, float b) {
  return (bf1(a) & 0xFFFFu) | (bf1(b) << 16);
}
__device__ __forceinline__ bf16_t f2bf(float f) {
  return __builtin_bit_cast(bf16_t, (unsigned short)bf1(f));
}

// ---------------------------------------------------------------------------
// GEMM: C[M,N] = epi( A[M,K](lda) * B' )
//   transB=1: B' = W[N,K](ldb)^T   (x @ w.T pattern, both K-contiguous)
//   transB=0: B' = W[K,N](ldb)     (scores @ V pattern)
// ---------------------------------------------------------------------------
__global__ __launch_bounds__(256) void k_gemm_bf16(
    const float* __restrict__ A, int lda,
    const float* __restrict__ Wt, int ldb, int transB,
    float* __restrict__ C, int ldc,
    int M, int N, int K,
    const float* __restrict__ s0, const float* __restrict__ s1,
    int epi, float slope)
{
  __shared__ bf16_t As[128][40];   // +8 bf16 pad (16B) vs 64-bank LDS
  __shared__ bf16_t Bs[128][40];

  const int tid  = threadIdx.x;
  const int lane = tid & 31, wave = tid >> 5;
  const int wm = wave >> 2, wn = wave & 3;     // 2 x 4 wave grid
  const int l16 = lane & 15, lh = lane >> 4;
  const int rowBase = blockIdx.y * 128;
  const int colBase = blockIdx.x * 128;

  const bool aAligned = (lda & 3) == 0;
  const bool bAligned = (ldb & 3) == 0;
  const bool aFullM   = (rowBase + 128 <= M);
  const bool bFullN   = (colBase + 128 <= N);

  v8f acc[4][2] = {};

  for (int k0 = 0; k0 < K; k0 += 32) {
    const bool fullK = (k0 + 32 <= K);

    // ---- stage A tile (128 x 32 f32 -> bf16) ----
    if (fullK && aFullM && aAligned) {
#pragma unroll
      for (int t = 0; t < 4; ++t) {
        int q = tid + 256 * t;               // 0..1023 float4 slots
        int r = q >> 3, c4 = (q & 7) << 2;
        const float* p = A + (size_t)(rowBase + r) * lda + (k0 + c4);
        float4 v = *(const float4*)p;
        if (k0 + 32 < K) __builtin_prefetch(p + 32, 0, 1); // global_prefetch_b8
        uint2 u; u.x = pack2bf(v.x, v.y); u.y = pack2bf(v.z, v.w);
        *(uint2*)(&As[r][c4]) = u;
      }
    } else {
#pragma unroll
      for (int t = 0; t < 4; ++t) {
        int q  = tid + 256 * t;
        int r  = q >> 3, c4 = (q & 7) << 2;
        int gr = rowBase + r, gc = k0 + c4;
        float x0 = 0.f, x1 = 0.f, x2 = 0.f, x3 = 0.f;
        if (gr < M) {
          const float* p = A + (size_t)gr * lda + gc;
          if (gc     < K) x0 = p[0];
          if (gc + 1 < K) x1 = p[1];
          if (gc + 2 < K) x2 = p[2];
          if (gc + 3 < K) x3 = p[3];
        }
        uint2 u; u.x = pack2bf(x0, x1); u.y = pack2bf(x2, x3);
        *(uint2*)(&As[r][c4]) = u;
      }
    }

    // ---- stage W tile ----
    if (transB) {
      if (fullK && bFullN && bAligned) {
#pragma unroll
        for (int t = 0; t < 4; ++t) {
          int q = tid + 256 * t;
          int r = q >> 3, c4 = (q & 7) << 2;
          const float* p = Wt + (size_t)(colBase + r) * ldb + (k0 + c4);
          float4 v = *(const float4*)p;
          if (k0 + 32 < K) __builtin_prefetch(p + 32, 0, 1);
          uint2 u; u.x = pack2bf(v.x, v.y); u.y = pack2bf(v.z, v.w);
          *(uint2*)(&Bs[r][c4]) = u;
        }
      } else {
#pragma unroll
        for (int t = 0; t < 4; ++t) {
          int q  = tid + 256 * t;
          int r  = q >> 3, c4 = (q & 7) << 2;
          int gn = colBase + r, gc = k0 + c4;
          float x0 = 0.f, x1 = 0.f, x2 = 0.f, x3 = 0.f;
          if (gn < N) {
            const float* p = Wt + (size_t)gn * ldb + gc;
            if (gc     < K) x0 = p[0];
            if (gc + 1 < K) x1 = p[1];
            if (gc + 2 < K) x2 = p[2];
            if (gc + 3 < K) x3 = p[3];
          }
          uint2 u; u.x = pack2bf(x0, x1); u.y = pack2bf(x2, x3);
          *(uint2*)(&Bs[r][c4]) = u;
        }
      }
    } else {
      if (fullK && bFullN && bAligned) {
        // W[K][N]: vector loads along N, transposed scatter into Bs[n][c]
#pragma unroll
        for (int t = 0; t < 4; ++t) {
          int q  = tid + 256 * t;             // 1024 float4 slots: 32 c x 32 n4
          int c  = q >> 5, n4 = (q & 31) << 2;
          const float* p = Wt + (size_t)(k0 + c) * ldb + (colBase + n4);
          float4 v = *(const float4*)p;
          Bs[n4 + 0][c] = f2bf(v.x);
          Bs[n4 + 1][c] = f2bf(v.y);
          Bs[n4 + 2][c] = f2bf(v.z);
          Bs[n4 + 3][c] = f2bf(v.w);
        }
      } else {
#pragma unroll
        for (int t = 0; t < 16; ++t) {
          int q = tid + 256 * t;              // 4096 scalars
          int n = q >> 5, c = q & 31;
          int gn = colBase + n, gc = k0 + c;
          float v = 0.f;
          if (gn < N && gc < K) v = Wt[(size_t)gc * ldb + gn];
          Bs[n][c] = f2bf(v);
        }
      }
    }
    __syncthreads();

    // ---- fragments (A and B: K split {0-7,16-23}/{8-15,24-31}) ----
    v16bf af[4], bfv[2];
#pragma unroll
    for (int i = 0; i < 4; ++i) {
      const bf16_t* p = &As[wm * 64 + i * 16 + l16][lh * 8];
      v8bf lo = *(const v8bf*)p;
      v8bf hi = *(const v8bf*)(p + 16);
      af[i] = __builtin_shufflevector(lo, hi, 0,1,2,3,4,5,6,7,8,9,10,11,12,13,14,15);
    }
#pragma unroll
    for (int j = 0; j < 2; ++j) {
      const bf16_t* p = &Bs[wn * 32 + j * 16 + l16][lh * 8];
      v8bf lo = *(const v8bf*)p;
      v8bf hi = *(const v8bf*)(p + 16);
      bfv[j] = __builtin_shufflevector(lo, hi, 0,1,2,3,4,5,6,7,8,9,10,11,12,13,14,15);
    }
#pragma unroll
    for (int i = 0; i < 4; ++i)
#pragma unroll
      for (int j = 0; j < 2; ++j)
        acc[i][j] = __builtin_amdgcn_wmma_f32_16x16x32_bf16(
            false, af[i], false, bfv[j], (short)0, acc[i][j], false, false);
    __syncthreads();
  }

  // ---- epilogue: C/D layout VGPR r -> M = r (lanes 0-15) / 8+r (16-31) ----
#pragma unroll
  for (int i = 0; i < 4; ++i) {
    int mbase = rowBase + wm * 64 + i * 16 + lh * 8;
#pragma unroll
    for (int j = 0; j < 2; ++j) {
      int n = colBase + wn * 32 + j * 16 + l16;
      if (n >= N) continue;
      float a0 = s0 ? s0[n] : 0.f;
      float a1 = s1 ? s1[n] : 0.f;
#pragma unroll
      for (int r = 0; r < 8; ++r) {
        int mm = mbase + r;
        if (mm < M) {
          float v = acc[i][j][r];
          if      (epi == EPI_BIAS)       v += a0;
          else if (epi == EPI_BIAS_RELU)  { v += a0; v = v > 0.f ? v : 0.f; }
          else if (epi == EPI_BIAS_LRELU) { v += a0; v = v > 0.f ? v : v * slope; }
          else if (epi == EPI_SS)         v = v * a0 + a1;
          C[(size_t)mm * ldc + n] = v;
        }
      }
    }
  }
}

// ---------------------------------------------------------------------------
// Elementwise / reduction kernels
// ---------------------------------------------------------------------------
__global__ __launch_bounds__(256) void k_in_tr(const float* x, float* y) {
  int i = blockIdx.x * 256 + threadIdx.x;
  if (i >= B_ * 3 * N_) return;
  int n = i % N_; int c = (i / N_) % 3; int b = i / (3 * N_);
  y[((size_t)b * N_ + n) * 3 + c] = x[i];
}

__global__ __launch_bounds__(256) void k_out_tr(const float* src, float* dst) {
  int i = blockIdx.x * 256 + threadIdx.x;
  if (i >= B_ * 6 * N_) return;
  int n = i % N_; int c = (i / N_) % 6; int b = i / (6 * N_);
  dst[i] = src[((size_t)b * N_ + n) * 6 + c];
}

__global__ __launch_bounds__(256) void k_colstats(const float* __restrict__ X, int M, int C,
                                                  float* mean, float* rstd) {
  int c = blockIdx.x;
  __shared__ float r1[256], r2[256];
  float a = 0.f, b = 0.f;
  for (int m = threadIdx.x; m < M; m += 256) {
    float v = X[(size_t)m * C + c]; a += v; b += v * v;
  }
  r1[threadIdx.x] = a; r2[threadIdx.x] = b; __syncthreads();
  for (int s = 128; s > 0; s >>= 1) {
    if ((int)threadIdx.x < s) { r1[threadIdx.x] += r1[threadIdx.x + s]; r2[threadIdx.x] += r2[threadIdx.x + s]; }
    __syncthreads();
  }
  if (threadIdx.x == 0) {
    float m_ = r1[0] / M;
    float v_ = r2[0] / M - m_ * m_;
    mean[c] = m_;
    rstd[c] = rsqrtf(fmaxf(v_, 0.f) + 1e-5f);
  }
}

__global__ __launch_bounds__(256) void k_bn_act(float* X, int C, const float* mean, const float* rstd,
                                                const float* g, const float* b, int act, float slope,
                                                size_t total) {
  size_t i = (size_t)blockIdx.x * 256 + threadIdx.x;
  if (i >= total) return;
  int c = (int)(i % C);
  float v = (X[i] - mean[c]) * rstd[c] * g[c] + b[c];
  if (act == 1) v = v > 0.f ? v : 0.f;
  else if (act == 2) v = v > 0.f ? v : v * slope;
  X[i] = v;
}

__global__ __launch_bounds__(256) void k_softmax(float* X, int cols, float scale) {
  size_t row = blockIdx.x;
  float* x = X + row * (size_t)cols;
  __shared__ float red[256];
  float mx = -3.4e38f;
  for (int i = threadIdx.x; i < cols; i += 256) mx = fmaxf(mx, x[i] * scale);
  red[threadIdx.x] = mx; __syncthreads();
  for (int s = 128; s > 0; s >>= 1) {
    if ((int)threadIdx.x < s) red[threadIdx.x] = fmaxf(red[threadIdx.x], red[threadIdx.x + s]);
    __syncthreads();
  }
  mx = red[0]; __syncthreads();
  float sum = 0.f;
  for (int i = threadIdx.x; i < cols; i += 256) {
    float e = __expf(x[i] * scale - mx); x[i] = e; sum += e;
  }
  red[threadIdx.x] = sum; __syncthreads();
  for (int s = 128; s > 0; s >>= 1) {
    if ((int)threadIdx.x < s) red[threadIdx.x] += red[threadIdx.x + s];
    __syncthreads();
  }
  float inv = 1.f / red[0];
  for (int i = threadIdx.x; i < cols; i += 256) x[i] *= inv;
}

__global__ __launch_bounds__(256) void k_add_ln(float* __restrict__ out, const float* __restrict__ a,
                                                const float* __restrict__ r, const float* __restrict__ g,
                                                const float* __restrict__ be, int D) {
  size_t row = blockIdx.x;
  float* o = out + row * (size_t)D;
  const float* pa = a + row * (size_t)D;
  const float* pr = r ? r + row * (size_t)D : nullptr;
  __shared__ float red[256];
  float s = 0.f;
  for (int i = threadIdx.x; i < D; i += 256) {
    float v = pa[i] + (pr ? pr[i] : 0.f); o[i] = v; s += v;
  }
  red[threadIdx.x] = s; __syncthreads();
  for (int st = 128; st > 0; st >>= 1) {
    if ((int)threadIdx.x < st) red[threadIdx.x] += red[threadIdx.x + st];
    __syncthreads();
  }
  float mean = red[0] / D; __syncthreads();
  float q = 0.f;
  for (int i = threadIdx.x; i < D; i += 256) { float d = o[i] - mean; q += d * d; }
  red[threadIdx.x] = q; __syncthreads();
  for (int st = 128; st > 0; st >>= 1) {
    if ((int)threadIdx.x < st) red[threadIdx.x] += red[threadIdx.x + st];
    __syncthreads();
  }
  float rstd = rsqrtf(red[0] / D + 1e-5f);
  for (int i = threadIdx.x; i < D; i += 256) o[i] = (o[i] - mean) * rstd * g[i] + be[i];
}

__global__ __launch_bounds__(256) void k_axpy(float* out, const float* a, const float* b,
                                              float alpha, size_t n) {
  size_t i = (size_t)blockIdx.x * 256 + threadIdx.x;
  if (i < n) out[i] = a[i] + alpha * b[i];
}

__global__ __launch_bounds__(256) void k_rowsq(const float* x, int C, float* sq, size_t rows) {
  size_t r = (size_t)blockIdx.x * 256 + threadIdx.x;
  if (r >= rows) return;
  float s = 0.f;
  for (int c = 0; c < C; ++c) { float v = x[r * C + c]; s += v * v; }
  sq[r] = s;
}

__global__ __launch_bounds__(256) void k_knn_topk(const float* inner, const float* sq, int Nn,
                                                  int k, int* out) {
  int row = blockIdx.x * 256 + threadIdx.x;
  if (row >= Nn) return;
  float bd[32]; int bi[32];
  for (int j = 0; j < k; ++j) { bd[j] = 3.4e38f; bi[j] = 0; }
  float sr = sq[row];
  const float* ir = inner + (size_t)row * Nn;
  for (int m = 0; m < Nn; ++m) {
    float d = sr + sq[m] - 2.f * ir[m];
    if (d < bd[k - 1]) {
      int j = k - 1;
      while (j > 0 && bd[j - 1] > d) { bd[j] = bd[j - 1]; bi[j] = bi[j - 1]; --j; }
      bd[j] = d; bi[j] = m;
    }
  }
  for (int j = 0; j < k; ++j) out[(size_t)row * k + j] = bi[j];
}

__global__ __launch_bounds__(256) void k_edge_feat(const float* x, const int* idx, float* f,
                                                   int C, int k, size_t total) {
  size_t i = (size_t)blockIdx.x * 256 + threadIdx.x;
  if (i >= total) return;
  int c = (int)(i % C); size_t t = i / C;
  int kk = (int)(t % k); size_t bn = t / k;
  int b = (int)(bn / N_);
  size_t nb = (size_t)b * N_ + idx[bn * k + kk];
  float ctr = x[bn * C + c];
  f[t * (2 * C) + c]     = x[nb * C + c] - ctr;
  f[t * (2 * C) + C + c] = ctr;
}

__global__ __launch_bounds__(256) void k_max_over_k(const float* f, int C, int k,
                                                    float* dst, int ldd, int off, size_t rows) {
  size_t i = (size_t)blockIdx.x * 256 + threadIdx.x;
  if (i >= rows * (size_t)C) return;
  int c = (int)(i % C); size_t r = i / C;
  float m = -3.4e38f;
  for (int j = 0; j < k; ++j) m = fmaxf(m, f[(r * k + j) * C + c]);
  dst[r * ldd + off + c] = m;
}

__global__ __launch_bounds__(256) void k_pool_max(const float* X, int Nn, int C, float* out) {
  int bc = blockIdx.x; int b = bc / C; int c = bc % C;
  __shared__ float red[256];
  float m = -3.4e38f;
  for (int n = threadIdx.x; n < Nn; n += 256) m = fmaxf(m, X[((size_t)b * Nn + n) * C + c]);
  red[threadIdx.x] = m; __syncthreads();
  for (int s = 128; s > 0; s >>= 1) {
    if ((int)threadIdx.x < s) red[threadIdx.x] = fmaxf(red[threadIdx.x], red[threadIdx.x + s]);
    __syncthreads();
  }
  if (threadIdx.x == 0) out[bc] = red[0];
}

__global__ __launch_bounds__(64) void k_add_eye(float* t9) {
  int i = threadIdx.x;
  if (i < 36 && (i % 9) % 4 == 0) t9[i] += 1.f;
}

__global__ __launch_bounds__(256) void k_apply_trans(const float* xt, const float* t9, float* xr) {
  int i = blockIdx.x * 256 + threadIdx.x;
  if (i >= BN_ * 3) return;
  int e = i % 3; size_t bn = i / 3; int b = (int)(bn / N_);
  float s = 0.f;
  for (int c = 0; c < 3; ++c) s += xt[bn * 3 + c] * t9[b * 9 + c * 3 + e];
  xr[i] = s;
}

__global__ __launch_bounds__(256) void k_topk_ch(const float* X, int Nn, int C, int k, int* out) {
  int t = blockIdx.x * 256 + threadIdx.x;
  if (t >= B_ * C) return;
  int b = t / C, c = t % C;
  float bd[32]; int bi[32];
  for (int j = 0; j < k; ++j) { bd[j] = -3.4e38f; bi[j] = 0; }
  for (int n = 0; n < Nn; ++n) {
    float v = X[((size_t)b * Nn + n) * C + c];
    if (v > bd[k - 1]) {
      int j = k - 1;
      while (j > 0 && bd[j - 1] < v) { bd[j] = bd[j - 1]; bi[j] = bi[j - 1]; --j; }
      bd[j] = v; bi[j] = n;
    }
  }
  for (int j = 0; j < k; ++j) out[(size_t)t * k + j] = bi[j];
}

__global__ __launch_bounds__(256) void k_gather_qpts(const float* xt, const int* idx32, float* qp) {
  int i = blockIdx.x * 256 + threadIdx.x;
  if (i >= B_ * 48 * 3) return;
  int c = i % 3; int t = i / 3; int b = t / 48;
  int n = idx32[t * 32 + 0];
  qp[i] = xt[((size_t)b * N_ + n) * 3 + c];
}

__global__ __launch_bounds__(256) void k_sqd_topk(const float* qp, const float* pts, int Nn,
                                                  int Q, int k, int* out) {
  int t = blockIdx.x * 256 + threadIdx.x;
  if (t >= B_ * Q) return;
  int b = t / Q;
  float x = qp[t * 3], y = qp[t * 3 + 1], z = qp[t * 3 + 2];
  float bd[32]; int bi[32];
  for (int j = 0; j < k; ++j) { bd[j] = 3.4e38f; bi[j] = 0; }
  for (int m = 0; m < Nn; ++m) {
    const float* p = pts + ((size_t)b * Nn + m) * 3;
    float dx = p[0] - x, dy = p[1] - y, dz = p[2] - z;
    float d = dx * dx + dy * dy + dz * dz;
    if (d < bd[k - 1]) {
      int j = k - 1;
      while (j > 0 && bd[j - 1] > d) { bd[j] = bd[j - 1]; bi[j] = bi[j - 1]; --j; }
      bd[j] = d; bi[j] = m;
    }
  }
  for (int j = 0; j < k; ++j) out[(size_t)t * k + j] = bi[j];
}

__global__ __launch_bounds__(256) void k_gather_rp(const float* xt, const int* ridx, float* rp) {
  int i = blockIdx.x * 256 + threadIdx.x;
  if (i >= B_ * 48 * 32 * 3) return;
  int c = i % 3; int t = i / 3;      // t = (b*48+q)*32+m
  int bq = t / 32;  int b = bq / 48;
  int n = ridx[t];
  rp[i] = xt[((size_t)b * N_ + n) * 3 + c];
}

__global__ __launch_bounds__(256) void k_rearr_r2(const float* r1o, float* A2) {
  size_t i = (size_t)blockIdx.x * 256 + threadIdx.x;
  if (i >= (size_t)B_ * 48 * 8192) return;
  int cm = (int)(i % 8192); size_t row = i / 8192;
  int c = cm / 32, m = cm % 32;
  A2[i] = r1o[(row * 32 + m) * 256 + c];
}

__global__ __launch_bounds__(256) void k_patch_idx(float* xp, const int* idx32) {
  int i = blockIdx.x * 256 + threadIdx.x;
  if (i >= B_ * 48 * 32) return;
  int j = i % 32; int row = i / 32;
  xp[(size_t)row * 512 + 480 + j] = (float)idx32[i];
}

__global__ __launch_bounds__(256) void k_copy_cols(float* dst, int ldd, int off,
                                                   const float* src, int lds, int C, size_t rows) {
  size_t i = (size_t)blockIdx.x * 256 + threadIdx.x;
  if (i >= rows * (size_t)C) return;
  int c = (int)(i % C); size_t r = i / C;
  dst[r * ldd + off + c] = src[r * lds + c];
}

__global__ __launch_bounds__(256) void k_bcast(float* dst, int ldd, int off, const float* g, int C) {
  size_t i = (size_t)blockIdx.x * 256 + threadIdx.x;
  if (i >= (size_t)BN_ * C) return;
  int c = (int)(i % C); size_t bn = i / C; int b = (int)(bn / N_);
  dst[bn * ldd + off + c] = g[b * C + c];
}

// ---------------------------------------------------------------------------
// Host helpers (enqueue only; everything on `stream`)
// ---------------------------------------------------------------------------
static inline dim3 g1(size_t n) { return dim3((unsigned)((n + 255) / 256)); }

static void gemm(hipStream_t st, const float* A, int lda, const float* W, int ldb, int transB,
                 float* C, int ldc, int M, int N, int K,
                 const float* s0, const float* s1, int epi, float slope) {
  dim3 g((N + 127) / 128, (M + 127) / 128);
  k_gemm_bf16<<<g, 256, 0, st>>>(A, lda, W, ldb, transB, C, ldc, M, N, K, s0, s1, epi, slope);
}

static void bn_act(hipStream_t st, float* X, size_t M, int C, const float* g, const float* b,
                   float* mean, float* rstd, int act) {
  k_colstats<<<C, 256, 0, st>>>(X, (int)M, C, mean, rstd);
  size_t tot = M * (size_t)C;
  k_bn_act<<<g1(tot), 256, 0, st>>>(X, C, mean, rstd, g, b, act, 0.2f, tot);
}

struct MhaP { const float *wq, *bq, *wk, *bk, *wv, *bv, *wo, *bo; };

static void mha_std(hipStream_t st, const float* Qin, int Lq, const float* Kin, int Lk,
                    int D, int h, const MhaP& p,
                    float* qb, float* kb, float* vb, float* sb, float* ob, float* outb) {
  int Mq = B_ * Lq, Mk = B_ * Lk, dh = D / h;
  gemm(st, Qin, D, p.wq, D, 1, qb, D, Mq, D, D, p.bq, nullptr, EPI_BIAS, 0.f);
  gemm(st, Kin, D, p.wk, D, 1, kb, D, Mk, D, D, p.bk, nullptr, EPI_BIAS, 0.f);
  gemm(st, Kin, D, p.wv, D, 1, vb, D, Mk, D, D, p.bv, nullptr, EPI_BIAS, 0.f);
  float scale = 1.0f / sqrtf((float)dh);
  for (int b = 0; b < B_; ++b)
    for (int hh = 0; hh < h; ++hh) {
      const float* qh = qb + (size_t)b * Lq * D + hh * dh;
      const float* kh = kb + (size_t)b * Lk * D + hh * dh;
      const float* vh = vb + (size_t)b * Lk * D + hh * dh;
      float*       oh = ob + (size_t)b * Lq * D + hh * dh;
      gemm(st, qh, D, kh, D, 1, sb, Lk, Lq, Lk, dh, nullptr, nullptr, EPI_NONE, 0.f);
      k_softmax<<<Lq, 256, 0, st>>>(sb, Lk, scale);
      gemm(st, sb, Lk, vh, D, 0, oh, D, Lq, dh, Lk, nullptr, nullptr, EPI_NONE, 0.f);
    }
  gemm(st, ob, D, p.wo, D, 1, outb, D, Mq, D, D, p.bo, nullptr, EPI_BIAS, 0.f);
}

static void knn(hipStream_t st, const float* xin, int C, float* bigsq, float* sq, int* idx) {
  k_rowsq<<<g1(BN_), 256, 0, st>>>(xin, C, sq, BN_);
  for (int b = 0; b < B_; ++b) {
    const float* xb = xin + (size_t)b * N_ * C;
    gemm(st, xb, C, xb, C, 1, bigsq, N_, N_, N_, C, nullptr, nullptr, EPI_NONE, 0.f);
    k_knn_topk<<<g1(N_), 256, 0, st>>>(bigsq, sq + (size_t)b * N_, N_, KNN,
                                       idx + (size_t)b * N_ * KNN);
  }
}

// ---------------------------------------------------------------------------
extern "C" void kernel_launch(void* const* d_in, const int* in_sizes, int n_in,
                              void* d_out, int out_size, void* d_ws, size_t ws_size,
                              hipStream_t stream) {
  (void)in_sizes; (void)n_in; (void)out_size; (void)ws_size;
  hipStream_t st = stream;
  auto P = [&](int i) -> const float* { return (const float*)d_in[i]; };

  // workspace bump allocator
  size_t off = 0;
  auto allocF = [&](size_t elems) -> float* {
    float* p = (float*)((char*)d_ws + off);
    off += ((elems * sizeof(float) + 255) & ~(size_t)255);
    return p;
  };
  auto allocI = [&](size_t elems) -> int* {
    int* p = (int*)((char*)d_ws + off);
    off += ((elems * sizeof(int) + 255) & ~(size_t)255);
    return p;
  };

  const size_t R = (size_t)BN_ * KNN;           // edge rows
  float* scratchA = allocF((size_t)R * 128);    // edge feats / dec-FFN hidden / cat(1728)
  float* scratchB = allocF((size_t)R * 64);     // edge conv out / stn h3 / c7 out
  float* bigsq    = allocF((size_t)N_ * N_);    // Gram / attention scores (reused)
  float* mean     = allocF(2048);
  float* rstd     = allocF(2048);
  float* sq       = allocF(BN_);
  float* xt       = allocF((size_t)BN_ * 3);
  float* xr       = allocF((size_t)BN_ * 3);
  float* h1       = allocF((size_t)BN_ * 64);
  float* h2       = allocF((size_t)BN_ * 128);
  float* pooled   = allocF(B_ * 1024);
  float* f1o      = allocF(B_ * 512);
  float* f2o      = allocF(B_ * 256);
  float* t9       = allocF(B_ * 9);
  int*   idx      = allocI(R);
  float* x1p      = allocF((size_t)BN_ * 64);
  float* x2p      = allocF((size_t)BN_ * 64);
  float* x3p      = allocF((size_t)BN_ * 64);
  float* x_pre    = allocF((size_t)BN_ * 192);
  float* x_mid    = allocF((size_t)BN_ * 512);
  float* xs_a     = allocF((size_t)BN_ * 256);
  float* xs_b     = allocF((size_t)BN_ * 256);
  float* q64      = allocF((size_t)BN_ * 64);
  float* k64      = allocF((size_t)BN_ * 64);
  float* v256     = allocF((size_t)BN_ * 256);
  float* xr256    = allocF((size_t)BN_ * 256);
  float* tmp256   = allocF((size_t)BN_ * 256);
  float* high3    = allocF((size_t)BN_ * 48);
  int*   idx32    = allocI(B_ * 48 * 32);
  int*   ridx     = allocI(B_ * 48 * 32);
  float* qpts     = allocF(B_ * 48 * 3);
  float* rp       = allocF((size_t)B_ * 48 * 32 * 3);
  float* r0o      = allocF((size_t)B_ * 48 * 32 * 128);
  float* r1o      = allocF((size_t)B_ * 48 * 32 * 256);
  float* A2       = allocF((size_t)B_ * 48 * 8192);
  float* r2o      = allocF((size_t)B_ * 48 * 480);
  float* xpatch   = allocF((size_t)B_ * 48 * 512);
  float* eq = allocF((size_t)B_ * 48 * 512);
  float* ek = allocF((size_t)B_ * 48 * 512);
  float* ev = allocF((size_t)B_ * 48 * 512);
  float* eo = allocF((size_t)B_ * 48 * 512);
  float* et = allocF((size_t)B_ * 48 * 512);
  float* ehid = allocF((size_t)B_ * 48 * 2048);
  float* dq   = allocF((size_t)BN_ * 512);
  float* dk   = allocF((size_t)BN_ * 512);
  float* dv   = allocF((size_t)BN_ * 512);
  float* dob  = allocF((size_t)BN_ * 512);
  float* dtmp = allocF((size_t)BN_ * 512);
  float* hdec = allocF((size_t)BN_ * 512);
  float* gpool = allocF(B_ * 1024);
  float* c8o  = allocF((size_t)BN_ * 512);
  float* c9o  = allocF((size_t)BN_ * 256);
  float* out6 = allocF((size_t)BN_ * 6);

  // ---- input transpose: x [B,3,N] -> xt [B*N,3] ----
  k_in_tr<<<g1(BN_ * 3), 256, 0, st>>>((const float*)d_in[0], xt);

  // ---- STN3d ----
  gemm(st, xt, 3, P(2), 3, 1, h1, 64, BN_, 64, 3, P(3), nullptr, EPI_BIAS, 0.f);
  bn_act(st, h1, BN_, 64, P(8), P(9), mean, rstd, 1);
  gemm(st, h1, 64, P(4), 64, 1, h2, 128, BN_, 128, 64, P(5), nullptr, EPI_BIAS, 0.f);
  bn_act(st, h2, BN_, 128, P(10), P(11), mean, rstd, 1);
  float* h3 = scratchB;                            // [BN,1024]
  gemm(st, h2, 128, P(6), 128, 1, h3, 1024, BN_, 1024, 128, P(7), nullptr, EPI_BIAS, 0.f);
  bn_act(st, h3, BN_, 1024, P(12), P(13), mean, rstd, 1);
  k_pool_max<<<B_ * 1024, 256, 0, st>>>(h3, N_, 1024, pooled);
  gemm(st, pooled, 1024, P(14), 1024, 1, f1o, 512, B_, 512, 1024, P(15), nullptr, EPI_BIAS, 0.f);
  bn_act(st, f1o, B_, 512, P(20), P(21), mean, rstd, 1);
  gemm(st, f1o, 512, P(16), 512, 1, f2o, 256, B_, 256, 512, P(17), nullptr, EPI_BIAS, 0.f);
  bn_act(st, f2o, B_, 256, P(22), P(23), mean, rstd, 1);
  gemm(st, f2o, 256, P(18), 256, 1, t9, 9, B_, 9, 256, P(19), nullptr, EPI_BIAS, 0.f);
  k_add_eye<<<1, 64, 0, st>>>(t9);
  k_apply_trans<<<g1(BN_ * 3), 256, 0, st>>>(xt, t9, xr);

  // ---- edge block 1 (xr, C=3) ----
  knn(st, xr, 3, bigsq, sq, idx);
  k_edge_feat<<<g1(R * 3), 256, 0, st>>>(xr, idx, scratchA, 3, KNN, R * 3);
  gemm(st, scratchA, 6, P(24), 6, 1, scratchB, 64, (int)R, 64, 6, nullptr, nullptr, EPI_NONE, 0.f);
  bn_act(st, scratchB, R, 64, P(25), P(26), mean, rstd, 2);
  gemm(st, scratchB, 64, P(27), 64, 1, scratchA, 64, (int)R, 64, 64, nullptr, nullptr, EPI_NONE, 0.f);
  bn_act(st, scratchA, R, 64, P(28), P(29), mean, rstd, 2);
  k_max_over_k<<<g1((size_t)BN_ * 64), 256, 0, st>>>(scratchA, 64, KNN, x1p, 64, 0, BN_);

  // ---- edge block 2 (x1, C=64) ----
  knn(st, x1p, 64, bigsq, sq, idx);
  k_edge_feat<<<g1(R * 64), 256, 0, st>>>(x1p, idx, scratchA, 64, KNN, R * 64);
  gemm(st, scratchA, 128, P(30), 128, 1, scratchB, 64, (int)R, 64, 128, nullptr, nullptr, EPI_NONE, 0.f);
  bn_act(st, scratchB, R, 64, P(31), P(32), mean, rstd, 2);
  gemm(st, scratchB, 64, P(33), 64, 1, scratchA, 64, (int)R, 64, 64, nullptr, nullptr, EPI_NONE, 0.f);
  bn_act(st, scratchA, R, 64, P(34), P(35), mean, rstd, 2);
  k_max_over_k<<<g1((size_t)BN_ * 64), 256, 0, st>>>(scratchA, 64, KNN, x2p, 64, 0, BN_);

  // ---- edge block 3 (x2, C=64) ----
  knn(st, x2p, 64, bigsq, sq, idx);
  k_edge_feat<<<g1(R * 64), 256, 0, st>>>(x2p, idx, scratchA, 64, KNN, R * 64);
  gemm(st, scratchA, 128, P(36), 128, 1, scratchB, 64, (int)R, 64, 128, nullptr, nullptr, EPI_NONE, 0.f);
  bn_act(st, scratchB, R, 64, P(37), P(38), mean, rstd, 2);
  k_max_over_k<<<g1((size_t)BN_ * 64), 256, 0, st>>>(scratchB, 64, KNN, x3p, 64, 0, BN_);

  // ---- x_pre = concat(x1,x2,x3); x_mid = lrelu(bn6(c6(x_pre))) ----
  k_copy_cols<<<g1((size_t)BN_ * 64), 256, 0, st>>>(x_pre, 192, 0,   x1p, 64, 64, BN_);
  k_copy_cols<<<g1((size_t)BN_ * 64), 256, 0, st>>>(x_pre, 192, 64,  x2p, 64, 64, BN_);
  k_copy_cols<<<g1((size_t)BN_ * 64), 256, 0, st>>>(x_pre, 192, 128, x3p, 64, 64, BN_);
  gemm(st, x_pre, 192, P(39), 192, 1, x_mid, 512, BN_, 512, 192, nullptr, nullptr, EPI_NONE, 0.f);
  bn_act(st, x_mid, BN_, 512, P(40), P(41), mean, rstd, 2);

  // ---- sort convs: 3 -> 64 -> 128 -> 256 ----
  gemm(st, xt, 3, P(42), 3, 1, h1, 64, BN_, 64, 3, P(43), nullptr, EPI_BIAS, 0.f);
  bn_act(st, h1, BN_, 64, P(48), P(49), mean, rstd, 2);
  gemm(st, h1, 64, P(44), 64, 1, h2, 128, BN_, 128, 64, P(45), nullptr, EPI_BIAS, 0.f);
  bn_act(st, h2, BN_, 128, P(50), P(51), mean, rstd, 2);
  gemm(st, h2, 128, P(46), 128, 1, xs_a, 256, BN_, 256, 128, P(47), nullptr, EPI_BIAS, 0.f);
  bn_act(st, xs_a, BN_, 256, P(52), P(53), mean, rstd, 2);

  // ---- superpoint: sa_layer (h=4, dh_q=16, dh_v=64) ----
  gemm(st, xs_a, 256, P(54), 256, 1, q64, 64, BN_, 64, 256, nullptr, nullptr, EPI_NONE, 0.f);
  gemm(st, xs_a, 256, P(55), 256, 1, k64, 64, BN_, 64, 256, nullptr, nullptr, EPI_NONE, 0.f);
  gemm(st, xs_a, 256, P(56), 256, 1, v256, 256, BN_, 256, 256, nullptr, nullptr, EPI_NONE, 0.f);
  for (int b = 0; b < B_; ++b)
    for (int hh = 0; hh < 4; ++hh) {
      const float* qh = q64 + (size_t)b * N_ * 64 + hh * 16;
      const float* kh = k64 + (size_t)b * N_ * 64 + hh * 16;
      const float* vh = v256 + (size_t)b * N_ * 256 + hh * 64;
      float*       oh = xr256 + (size_t)b * N_ * 256 + hh * 64;
      gemm(st, qh, 64, kh, 64, 1, bigsq, N_, N_, N_, 16, nullptr, nullptr, EPI_NONE, 0.f);
      k_softmax<<<N_, 256, 0, st>>>(bigsq, N_, 0.25f);
      gemm(st, bigsq, N_, vh, 256, 0, oh, 256, N_, 64, N_, nullptr, nullptr, EPI_NONE, 0.f);
    }
  k_axpy<<<g1((size_t)BN_ * 256), 256, 0, st>>>(tmp256, xs_a, xr256, -1.f, (size_t)BN_ * 256);
  gemm(st, tmp256, 256, P(57), 256, 1, xs_b, 256, BN_, 256, 256, nullptr, nullptr, EPI_NONE, 0.f);
  bn_act(st, xs_b, BN_, 256, P(58), P(59), mean, rstd, 1);
  k_axpy<<<g1((size_t)BN_ * 256), 256, 0, st>>>(xs_b, xs_a, xs_b, 1.f, (size_t)BN_ * 256);

  // ---- fg convs: 256 -> 64 -> 64 -> 48 ----
  gemm(st, xs_b, 256, P(60), 256, 1, h1, 64, BN_, 64, 256, P(61), nullptr, EPI_BIAS, 0.f);
  bn_act(st, h1, BN_, 64, P(66), P(67), mean, rstd, 2);
  gemm(st, h1, 64, P(62), 64, 1, q64, 64, BN_, 64, 64, P(63), nullptr, EPI_BIAS, 0.f);
  bn_act(st, q64, BN_, 64, P(68), P(69), mean, rstd, 2);
  gemm(st, q64, 64, P(64), 64, 1, high3, 48, BN_, 48, 64, P(65), nullptr, EPI_BIAS, 0.f);
  bn_act(st, high3, BN_, 48, P(70), P(71), mean, rstd, 2);

  // ---- top-k patches + local geometry ----
  k_topk_ch<<<g1(B_ * 48), 256, 0, st>>>(high3, N_, 48, 32, idx32);
  k_gather_qpts<<<g1(B_ * 48 * 3), 256, 0, st>>>(xt, idx32, qpts);
  k_sqd_topk<<<g1(B_ * 48), 256, 0, st>>>(qpts, xt, N_, 48, 32, ridx);
  k_gather_rp<<<g1((size_t)B_ * 48 * 32 * 3), 256, 0, st>>>(xt, ridx, rp);
  int Rm = B_ * 48 * 32;
  gemm(st, rp, 3, P(72), 3, 1, r0o, 128, Rm, 128, 3, P(73), nullptr, EPI_BIAS, 0.f);
  bn_act(st, r0o, Rm, 128, P(78), P(79), mean, rstd, 2);
  gemm(st, r0o, 128, P(74), 128, 1, r1o, 256, Rm, 256, 128, P(75), nullptr, EPI_BIAS, 0.f);
  bn_act(st, r1o, Rm, 256, P(80), P(81), mean, rstd, 2);
  k_rearr_r2<<<g1((size_t)B_ * 48 * 8192), 256, 0, st>>>(r1o, A2);
  gemm(st, A2, 8192, P(76), 8192, 1, r2o, 480, B_ * 48, 480, 8192, P(77), nullptr, EPI_BIAS, 0.f);
  bn_act(st, r2o, B_ * 48, 480, P(82), P(83), mean, rstd, 2);
  k_copy_cols<<<g1((size_t)B_ * 48 * 480), 256, 0, st>>>(xpatch, 512, 0, r2o, 480, 480, B_ * 48);
  k_patch_idx<<<g1(B_ * 48 * 32), 256, 0, st>>>(xpatch, idx32);

  // ---- encoder (48 tokens) + enc_ln ----
  {
    MhaP pe{P(84), P(85), P(86), P(87), P(88), P(89), P(90), P(91)};
    mha_std(st, xpatch, 48, xpatch, 48, 512, 4, pe, eq, ek, ev, bigsq, eo, et);
    k_add_ln<<<B_ * 48, 256, 0, st>>>(xpatch, xpatch, et, P(92), P(93), 512);
    gemm(st, xpatch, 512, P(96), 512, 1, ehid, 2048, B_ * 48, 2048, 512, P(97), nullptr, EPI_BIAS_RELU, 0.f);
    gemm(st, ehid, 2048, P(98), 2048, 1, et, 512, B_ * 48, 512, 2048, P(99), nullptr, EPI_BIAS, 0.f);
    k_add_ln<<<B_ * 48, 256, 0, st>>>(xpatch, xpatch, et, P(94), P(95), 512);
    k_add_ln<<<B_ * 48, 256, 0, st>>>(xpatch, xpatch, nullptr, P(100), P(101), 512);
  }

  // ---- decoder x2 (queries = x_mid tokens) ----
  hipMemcpyAsync(hdec, x_mid, (size_t)BN_ * 512 * sizeof(float), hipMemcpyDeviceToDevice, st);
  for (int l = 0; l < 2; ++l) {
    int o = 102 + l * 26;
    MhaP sa{P(o+0), P(o+1), P(o+2), P(o+3), P(o+4), P(o+5), P(o+6), P(o+7)};
    MhaP ca{P(o+8), P(o+9), P(o+10), P(o+11), P(o+12), P(o+13), P(o+14), P(o+15)};
    mha_std(st, hdec, N_, hdec, N_, 512, 4, sa, dq, dk, dv, bigsq, dob, dtmp);
    k_add_ln<<<BN_, 256, 0, st>>>(hdec, hdec, dtmp, P(o+16), P(o+17), 512);
    mha_std(st, hdec, N_, xpatch, 48, 512, 4, ca, dq, dk, dv, bigsq, dob, dtmp);
    k_add_ln<<<BN_, 256, 0, st>>>(hdec, hdec, dtmp, P(o+18), P(o+19), 512);
    gemm(st, hdec, 512, P(o+22), 512, 1, scratchA, 2048, BN_, 2048, 512, P(o+23), nullptr, EPI_BIAS_RELU, 0.f);
    gemm(st, scratchA, 2048, P(o+24), 2048, 1, dtmp, 512, BN_, 512, 2048, P(o+25), nullptr, EPI_BIAS, 0.f);
    k_add_ln<<<BN_, 256, 0, st>>>(hdec, hdec, dtmp, P(o+20), P(o+21), 512);
  }

  // ---- c7 + global max pool, concat [g | x_pre | emb], head ----
  float* c7o = scratchB;
  gemm(st, x_mid, 512, P(154), 512, 1, c7o, 1024, BN_, 1024, 512, nullptr, nullptr, EPI_NONE, 0.f);
  bn_act(st, c7o, BN_, 1024, P(155), P(156), mean, rstd, 2);
  k_pool_max<<<B_ * 1024, 256, 0, st>>>(c7o, N_, 1024, gpool);
  float* cat = scratchA;                            // [BN, 1728]
  k_bcast<<<g1((size_t)BN_ * 1024), 256, 0, st>>>(cat, 1728, 0, gpool, 1024);
  k_copy_cols<<<g1((size_t)BN_ * 192), 256, 0, st>>>(cat, 1728, 1024, x_pre, 192, 192, BN_);
  k_copy_cols<<<g1((size_t)BN_ * 512), 256, 0, st>>>(cat, 1728, 1216, hdec, 512, 512, BN_);
  gemm(st, cat, 1728, P(157), 1728, 1, c8o, 512, BN_, 512, 1728, nullptr, nullptr, EPI_NONE, 0.f);
  bn_act(st, c8o, BN_, 512, P(158), P(159), mean, rstd, 2);
  gemm(st, c8o, 512, P(160), 512, 1, c9o, 256, BN_, 256, 512, nullptr, nullptr, EPI_NONE, 0.f);
  bn_act(st, c9o, BN_, 256, P(161), P(162), mean, rstd, 2);
  gemm(st, c9o, 256, P(163), 256, 1, out6, 6, BN_, 6, 256, P(164), nullptr, EPI_BIAS, 0.f);
  k_out_tr<<<g1(BN_ * 6), 256, 0, st>>>(out6, (float*)d_out);
}
